// MHAttention_12103217840331
// MI455X (gfx1250) — matrix operand order repaired and checked by reading, not compiled, not run on hardware
//
#include <hip/hip_runtime.h>
#include <hip/hip_bf16.h>

// Problem constants (from reference): N=2048, B=2, C=1024, H=16, Dh=64
#define N_   2048
#define B_   2
#define C_   1024
#define H_   16
#define D_   64
#define NB_  (N_ * B_)       // 4096 flattened (n,b) rows

typedef __attribute__((ext_vector_type(16))) __bf16 v16bf;
typedef __attribute__((ext_vector_type(8)))  float  v8f;
typedef __attribute__((ext_vector_type(4)))  float  vf4;
typedef __attribute__((ext_vector_type(2)))  float  vf2;
typedef __attribute__((ext_vector_type(2)))  unsigned vu2;

union FragAB { v16bf v; unsigned u[8]; };
union FragC  { v8f   v; float    f[8]; };

__device__ __forceinline__ unsigned short f2bf(float f) {
    unsigned u = __float_as_uint(f);
    u += 0x7FFFu + ((u >> 16) & 1u);   // round-to-nearest-even
    return (unsigned short)(u >> 16);
}
__device__ __forceinline__ unsigned pack2bf(float a, float b) {
    return (unsigned)f2bf(a) | ((unsigned)f2bf(b) << 16);
}

// --- CDNA5 async global->LDS copy (16B per lane), tracked by ASYNCcnt -------
__device__ __forceinline__ void async_ld16(unsigned lds_off, const void* gaddr) {
    asm volatile("global_load_async_to_lds_b128 %0, %1, off"
                 :: "v"(lds_off), "v"(gaddr) : "memory");
}
__device__ __forceinline__ void wait_async0() { asm volatile("s_wait_asynccnt 0x0" ::: "memory"); }
__device__ __forceinline__ void wait_async1() { asm volatile("s_wait_asynccnt 0x1" ::: "memory"); }
__device__ __forceinline__ void wait_async2() { asm volatile("s_wait_asynccnt 0x2" ::: "memory"); }

// ---------------------------------------------------------------------------
// Projection: Y = X @ W^T, head-split to (B,H,N,Dh) bf16, optional L2 norm.
// Grid: (NB/32, H). Block: 256 threads = 8 waves; each wave owns one 16x16
// tile of the 32x64 (rows x head-dim) output strip. Staging uses 16B global
// loads + 8B LDS stores (no per-element serialization).
// ---------------------------------------------------------------------------
__global__ void proj_bf16_kernel(const float* __restrict__ X,
                                 const float* __restrict__ W,
                                 unsigned short* __restrict__ out,
                                 int do_norm)
{
    __shared__ __align__(16) unsigned short atile[32][32];  // X tile bf16 [row][k]
    __shared__ __align__(16) unsigned short wtile[64][32];  // W tile bf16 [c][k]
    __shared__ __align__(16) float          stile[32][64];  // f32 result strip
    __shared__ float                        rscale[32];

    const int tid  = threadIdx.x;
    const int lane = tid & 31;
    const int wave = tid >> 5;
    const int mi   = wave >> 2;
    const int ni   = wave & 3;
    const int rb   = blockIdx.x;
    const int h    = blockIdx.y;
    const int lm   = lane & 15;
    const int hi   = (lane >> 4) & 1;

    // staging coordinates (straight-line, exact counts)
    const int ar = tid >> 3, akq = (tid & 7) * 4;          // A: 256 x 16B chunks
    const int wr0 = tid >> 3, wr1 = (tid + 256) >> 3;      // W: 512 x 16B chunks

    FragC acc;
    #pragma unroll
    for (int r = 0; r < 8; ++r) acc.f[r] = 0.0f;

    const float* Xrow  = X + (size_t)(rb * 32 + ar) * C_ + akq;
    const float* Wrow0 = W + (size_t)(h * 64 + wr0) * C_ + akq;
    const float* Wrow1 = W + (size_t)(h * 64 + wr1) * C_ + akq;

    for (int k0 = 0; k0 < C_; k0 += 32) {
        vf4 xa = *(const vf4*)(Xrow + k0);
        vf4 w0 = *(const vf4*)(Wrow0 + k0);
        vf4 w1 = *(const vf4*)(Wrow1 + k0);
        vu2 pa; pa.x = pack2bf(xa.x, xa.y); pa.y = pack2bf(xa.z, xa.w);
        vu2 p0; p0.x = pack2bf(w0.x, w0.y); p0.y = pack2bf(w0.z, w0.w);
        vu2 p1; p1.x = pack2bf(w1.x, w1.y); p1.y = pack2bf(w1.z, w1.w);
        *(vu2*)&atile[ar][akq]  = pa;
        *(vu2*)&wtile[wr0][akq] = p0;
        *(vu2*)&wtile[wr1][akq] = p1;
        __syncthreads();

        FragAB a, bg;
        #pragma unroll
        for (int v2 = 0; v2 < 8; ++v2) {
            int ka = 2 * v2 + hi * 8 + ((v2 >= 4) ? 8 : 0);       // A layout
            a.u[v2]  = *(const unsigned*)&atile[mi * 16 + lm][ka];
            int kb = 2 * v2 + hi * 8 + ((v2 >= 4) ? 8 : 0);       // B layout
            bg.u[v2] = *(const unsigned*)&wtile[ni * 16 + lm][kb];
        }
        acc.v = __builtin_amdgcn_wmma_f32_16x16x32_bf16(
            false, a.v, false, bg.v, (short)0, acc.v, false, false);
        __syncthreads();
    }

    #pragma unroll
    for (int r = 0; r < 8; ++r) {
        stile[mi * 16 + r + hi * 8][ni * 16 + lm] = acc.f[r];
    }
    __syncthreads();

    if (do_norm && tid < 32) {
        float s = 0.0f;
        #pragma unroll 8
        for (int d = 0; d < 64; ++d) { float x = stile[tid][d]; s += x * x; }
        rscale[tid] = rsqrtf(s);
    }
    __syncthreads();

    // write bf16 to (B,H,N,Dh): 512 x (4 cols) chunks, 2 per thread
    #pragma unroll
    for (int j = 0; j < 2; ++j) {
        int i = tid + j * 256;
        int r = i >> 4, dq = (i & 15) * 4;
        int gr = rb * 32 + r;          // gr = n*B + b
        int n  = gr >> 1;              // B_ == 2
        int bb = gr & 1;
        vf4 val = *(const vf4*)&stile[r][dq];
        if (do_norm) {
            float s = rscale[r];
            val.x *= s; val.y *= s; val.z *= s; val.w *= s;
        }
        vu2 p; p.x = pack2bf(val.x, val.y); p.y = pack2bf(val.z, val.w);
        *(vu2*)&out[(((size_t)(bb * H_ + h)) * N_ + n) * D_ + dq] = p;
    }
}

// ---------------------------------------------------------------------------
// Attention: one workgroup owns 32 query rows of one (b,h).
// 32x2048 f32 score strip lives in LDS (256KB of the 320KB WGP LDS).
// K/V tiles stream in via global_load_async_to_lds_b128 with double
// buffering and partial s_wait_asynccnt waits.
// Grid: (N/32, H, B). Block: 256 threads = 8 waves.
// ---------------------------------------------------------------------------
#define SMEM_S     0
#define SMEM_Q     (32 * N_ * 4)             // 262144
#define SMEM_KV0   (SMEM_Q + 32 * 64 * 2)    // 266240
#define SMEM_KV1   (SMEM_KV0 + 8192)         // 274432
#define SMEM_RED   (SMEM_KV1 + 8192)         // 282624
#define SMEM_SCALE (SMEM_RED + 32 * 8 * 4)   // 283648
#define SMEM_TOTAL (SMEM_SCALE + 32 * 4)     // 283776

__global__ void cosattn_kernel(const unsigned short* __restrict__ qws,
                               const unsigned short* __restrict__ kws,
                               const unsigned short* __restrict__ vws,
                               float* __restrict__ attn_out,
                               float* __restrict__ x_out)
{
    extern __shared__ char smem_raw[];
    float*          S        = (float*)(smem_raw + SMEM_S);         // [32][2048]
    unsigned short* qtile    = (unsigned short*)(smem_raw + SMEM_Q);
    unsigned short* kvt0p    = (unsigned short*)(smem_raw + SMEM_KV0);
    unsigned short* kvt1p    = (unsigned short*)(smem_raw + SMEM_KV1);
    float*          red      = (float*)(smem_raw + SMEM_RED);       // [32][8]
    float*          rowscale = (float*)(smem_raw + SMEM_SCALE);     // [32]

    // LDS byte offsets for async copies (addr[31:0] of the flat LDS address);
    // keep these as scalars so no constant-addrspacecast aggregate is emitted.
    unsigned smbase = (unsigned)(size_t)(void*)smem_raw;
    unsigned qoff   = smbase + SMEM_Q;
    unsigned kvoff0 = smbase + SMEM_KV0;
    unsigned kvoff1 = smbase + SMEM_KV1;

    const int tid  = threadIdx.x;
    const int lane = tid & 31;
    const int wave = tid >> 5;
    const int mi   = wave >> 2;
    const int ni   = wave & 3;
    const int rb   = blockIdx.x;
    const int h    = blockIdx.y;
    const int b    = blockIdx.z;
    const int lm   = lane & 15;
    const int hi   = (lane >> 4) & 1;

    const size_t headbase = ((size_t)(b * H_ + h)) * N_ * D_;
    const char* kbase = (const char*)(kws + headbase);
    const char* vbase = (const char*)(vws + headbase);

    // --- async stage: q tile (4KB, 1 chunk/thread) + k tile 0 (8KB, 2/thread)
    {
        const char* qsrc = (const char*)(qws + headbase + (size_t)rb * 32 * D_);
        async_ld16(qoff + tid * 16, qsrc + tid * 16);
        async_ld16(kvoff0 + tid * 16,        kbase + tid * 16);
        async_ld16(kvoff0 + tid * 16 + 4096, kbase + tid * 16 + 4096);
    }
    wait_async0();
    __syncthreads();

    // hoist q fragments (kt-invariant): kc = 0 and kc = 32
    FragAB a0, a1;
    #pragma unroll
    for (int v2 = 0; v2 < 8; ++v2) {
        int ka = 2 * v2 + hi * 8 + ((v2 >= 4) ? 8 : 0);
        a0.u[v2] = *(const unsigned*)&qtile[(mi * 16 + lm) * D_ + ka];
        a1.u[v2] = *(const unsigned*)&qtile[(mi * 16 + lm) * D_ + 32 + ka];
    }

    // ---- phase 1: S = q k^T, double-buffered k tiles ----
    for (int it = 0; it < N_ / 64; ++it) {
        const int kt  = it * 64;
        const int cur = it & 1;
        const unsigned kvo_nxt = cur ? kvoff0 : kvoff1;
        if (it + 1 < N_ / 64) {
            const char* src = kbase + (size_t)(kt + 64) * (D_ * 2);
            async_ld16(kvo_nxt + tid * 16,        src + tid * 16);
            async_ld16(kvo_nxt + tid * 16 + 4096, src + tid * 16 + 4096);
            wait_async2();            // current tile's 2 chunks retired
        } else {
            wait_async0();
        }
        __syncthreads();

        const unsigned short* kvc = cur ? kvt1p : kvt0p;
        FragC acc;
        #pragma unroll
        for (int r = 0; r < 8; ++r) acc.f[r] = 0.0f;

        FragAB bg;
        #pragma unroll
        for (int v2 = 0; v2 < 8; ++v2) {
            int kb = 2 * v2 + hi * 8 + ((v2 >= 4) ? 8 : 0);
            bg.u[v2] = *(const unsigned*)&kvc[(ni * 16 + lm) * D_ + kb];
        }
        acc.v = __builtin_amdgcn_wmma_f32_16x16x32_bf16(
            false, a0.v, false, bg.v, (short)0, acc.v, false, false);
        #pragma unroll
        for (int v2 = 0; v2 < 8; ++v2) {
            int kb = 32 + 2 * v2 + hi * 8 + ((v2 >= 4) ? 8 : 0);
            bg.u[v2] = *(const unsigned*)&kvc[(ni * 16 + lm) * D_ + kb];
        }
        acc.v = __builtin_amdgcn_wmma_f32_16x16x32_bf16(
            false, a1.v, false, bg.v, (short)0, acc.v, false, false);

        #pragma unroll
        for (int r = 0; r < 8; ++r) {
            S[(mi * 16 + r + hi * 8) * N_ + kt + ni * 16 + lm] = acc.f[r];
        }
        __syncthreads();
    }

    // overlap: kick off v tile 0 (4KB) while softmax runs
    async_ld16(kvoff0 + tid * 16, vbase + tid * 16);

    // ---- phase 2: softmax (cosine scores in [-1,1] -> no max-shift) ----
    {
        int row = tid >> 3;            // 8 threads per row, 512 f4 chunks/row
        int sub = tid & 7;
        vf4* Srow = (vf4*)(S + row * N_);
        float psum = 0.0f;
        for (int c = sub; c < N_ / 4; c += 8) {
            vf4 v = Srow[c];
            vf4 e;
            e.x = __expf(v.x); e.y = __expf(v.y);
            e.z = __expf(v.z); e.w = __expf(v.w);
            Srow[c] = e;
            psum += (e.x + e.y) + (e.z + e.w);
        }
        red[row * 8 + sub] = psum;
    }
    __syncthreads();
    if (tid < 32) {
        float s = 0.0f;
        #pragma unroll
        for (int j = 0; j < 8; ++j) s += red[tid * 8 + j];
        rowscale[tid] = 1.0f / s;
    }
    __syncthreads();
    {
        vf4* S4 = (vf4*)S;
        vf4* aout4 = (vf4*)(attn_out + ((size_t)(b * H_ + h) * N_ + (size_t)rb * 32) * N_);
        for (int i = tid; i < 32 * (N_ / 4); i += 256) {
            int row = i >> 9;          // / (2048/4)
            float s = rowscale[row];
            vf4 p = S4[i];
            p.x *= s; p.y *= s; p.z *= s; p.w *= s;
            S4[i] = p;
            __builtin_nontemporal_store(p, &aout4[i]);  // 537MB stream, never re-read
        }
    }
    __syncthreads();

    // ---- phase 3: x = P @ v, double-buffered v tiles ----
    FragC xacc;
    #pragma unroll
    for (int r = 0; r < 8; ++r) xacc.f[r] = 0.0f;

    for (int it = 0; it < N_ / 32; ++it) {
        const int kt  = it * 32;
        const int cur = it & 1;
        const unsigned kvo_nxt = cur ? kvoff0 : kvoff1;
        if (it + 1 < N_ / 32) {
            async_ld16(kvo_nxt + tid * 16,
                       vbase + (size_t)(kt + 32) * (D_ * 2) + tid * 16);
            wait_async1();            // current tile's chunk retired
        } else {
            wait_async0();
        }
        __syncthreads();

        const unsigned short* vt = cur ? kvt1p : kvt0p;
        FragAB a, bg;
        #pragma unroll
        for (int v2 = 0; v2 < 8; ++v2) {
            int ka = kt + 2 * v2 + hi * 8 + ((v2 >= 4) ? 8 : 0);
            vf2 pf = *(const vf2*)&S[(mi * 16 + lm) * N_ + ka];
            a.u[v2] = pack2bf(pf.x, pf.y);
            int kb = 2 * v2 + hi * 8 + ((v2 >= 4) ? 8 : 0); // local key row in vtile
            unsigned short e0 = vt[(kb    ) * D_ + ni * 16 + lm];
            unsigned short e1 = vt[(kb + 1) * D_ + ni * 16 + lm];
            bg.u[v2] = (unsigned)e0 | ((unsigned)e1 << 16);
        }
        xacc.v = __builtin_amdgcn_wmma_f32_16x16x32_bf16(
            false, a.v, false, bg.v, (short)0, xacc.v, false, false);
        __syncthreads();
    }

    // write x: (N,B,C) with c = h*64 + d
    #pragma unroll
    for (int r = 0; r < 8; ++r) {
        int n = rb * 32 + mi * 16 + r + hi * 8;
        int c = h * 64 + ni * 16 + lm;
        x_out[((size_t)n * B_ + b) * C_ + c] = xacc.f[r];
    }
}

// ---------------------------------------------------------------------------
extern "C" void kernel_launch(void* const* d_in, const int* in_sizes, int n_in,
                              void* d_out, int out_size, void* d_ws, size_t ws_size,
                              hipStream_t stream) {
    (void)in_sizes; (void)n_in; (void)out_size; (void)ws_size;

    const float* Xq = (const float*)d_in[0];
    const float* Xk = (const float*)d_in[1];
    const float* Xv = (const float*)d_in[2];
    const float* Wq = (const float*)d_in[3];
    const float* Wk = (const float*)d_in[4];
    const float* Wv = (const float*)d_in[5];

    float* x_out    = (float*)d_out;                               // (N,B,C)
    float* attn_out = x_out + (size_t)N_ * B_ * C_;                // (B,H,N,N)

    const size_t headsz = (size_t)B_ * H_ * N_ * D_;               // elems per tensor
    unsigned short* qws = (unsigned short*)d_ws;
    unsigned short* kws = qws + headsz;
    unsigned short* vws = kws + headsz;

    dim3 gp(NB_ / 32, H_);
    proj_bf16_kernel<<<gp, 256, 0, stream>>>(Xq, Wq, qws, 1);
    proj_bf16_kernel<<<gp, 256, 0, stream>>>(Xk, Wk, kws, 1);
    proj_bf16_kernel<<<gp, 256, 0, stream>>>(Xv, Wv, vws, 0);

    dim3 ga(N_ / 32, H_, B_);
    cosattn_kernel<<<ga, 256, SMEM_TOTAL, stream>>>(qws, kws, vws, attn_out, x_out);
}
